// GaussianImage_Cholesky_HSI_85847806312499
// MI455X (gfx1250) — matrix operand order repaired and checked by reading, not compiled, not run on hardware
//
#include <hip/hip_runtime.h>
#include <hip/hip_bf16.h>

typedef __attribute__((ext_vector_type(16))) _Float16 v16h;
typedef __attribute__((ext_vector_type(8)))  float    v8f;

#define N_G   1024
#define WDIM  256
#define HDIM  256
#define NPIX  (WDIM * HDIM)     // 65536
#define NCH   64
#define NCHUNK (N_G / 32)       // 32 K-chunks of 32 gaussians

#define LOG2E   1.4426950408889634f
#define INV2PI  0.15915494309189535f

union AOp { v16h v; _Float16 e[16]; };

// ---------------------------------------------------------------------------
// Kernel 1: per-gaussian preprocessing -> AoS of 12 floats (48 B) per gaussian
//   q0 = [xs, ys, ca_s, cb_s]   (conic pre-scaled by -log2e, halves folded)
//   q1 = [cc_s, amp, f0x, f0y]  (gabor freqs pre-scaled by 1/2pi)
//   q2 = [f1x, f1y, w0, w1]
// so that  exp(-sigma) = exp2(ca_s*dx^2 + cc_s*dy^2 + cb_s*dx*dy)
// and      cos(fx*dx+fy*dy) = v_cos(f'x*dx + f'y*dy)  directly.
// ---------------------------------------------------------------------------
__global__ __launch_bounds__(256) void gi_precompute(
    const float* __restrict__ xyz,  const float* __restrict__ chol,
    const float* __restrict__ opac, const float* __restrict__ gfreq,
    const float* __restrict__ gwt,  float* __restrict__ P)
{
  int i = blockIdx.x * blockDim.x + threadIdx.x;
  if (i >= N_G) return;

  float tx = tanhf(xyz[2 * i + 0]);
  float ty = tanhf(xyz[2 * i + 1]);
  float xs = 0.5f * (float)WDIM * (tx + 1.0f);
  float ys = 0.5f * (float)HDIM * (ty + 1.0f);

  float l1 = chol[3 * i + 0] + 0.5f;
  float l2 = chol[3 * i + 1] + 0.0f;
  float l3 = chol[3 * i + 2] + 0.5f;
  float s11 = l1 * l1;
  float s12 = l1 * l2;
  float s22 = l2 * l2 + l3 * l3;
  float det = (l1 * l3) * (l1 * l3);
  float inv = 1.0f / det;
  float ca =  s22 * inv;
  float cb = -s12 * inv;
  float cc =  s11 * inv;

  float mid = 0.5f * (s11 + s22);
  float lam = mid + sqrtf(fmaxf(mid * mid - det, 1e-8f));
  float radius = ceilf(3.0f * sqrtf(lam));
  float keep = (radius > 1.0f) ? 1.0f : 0.0f;
  float amp = opac[i] * keep;

  // fold -log2e (and the 0.5) into the conic -> exp2() argument directly
  float ca_s = -0.5f * LOG2E * ca;
  float cb_s = -1.0f * LOG2E * cb;
  float cc_s = -0.5f * LOG2E * cc;

  // gabor_freqs reshaped (N, K=2, 2): idx [i,k,d] = (2i+k)*2 + d
  // fold 1/2pi -> v_cos_f32 operates on revolutions
  float f0x = __expf(gfreq[(2 * i + 0) * 2 + 0]) * INV2PI;
  float f0y = __expf(gfreq[(2 * i + 0) * 2 + 1]) * INV2PI;
  float f1x = __expf(gfreq[(2 * i + 1) * 2 + 0]) * INV2PI;
  float f1y = __expf(gfreq[(2 * i + 1) * 2 + 1]) * INV2PI;
  float w0 = 1.0f / (1.0f + __expf(-gwt[2 * i + 0]));
  float w1 = 1.0f / (1.0f + __expf(-gwt[2 * i + 1]));

  float* q = P + i * 12;
  q[0] = xs;   q[1] = ys;   q[2]  = ca_s; q[3]  = cb_s;
  q[4] = cc_s; q[5] = amp;  q[6]  = f0x;  q[7]  = f0y;
  q[8] = f1x;  q[9] = f1y;  q[10] = w0;   q[11] = w1;
}

// ---------------------------------------------------------------------------
// Kernel 2: softplus(features_dc) -> f16, pre-shuffled into dense WMMA B
// layout (32x16 f16 per chunk/channel-tile):
//   lanes 0-15 hold K={0..7,16..23} (2 K per VGPR), lanes 16-31 hold K={8..15,24..31},
//   column N = lane & 15.  Entry t = ((kc*4 + ct)*32 + lane)*16 + j.
// ---------------------------------------------------------------------------
__global__ __launch_bounds__(256) void gi_packB(
    const float* __restrict__ feat, _Float16* __restrict__ bpack)
{
  int t = blockIdx.x * 256 + threadIdx.x;         // 0 .. 65535
  int j    =  t        & 15;
  int lane = (t >> 4)  & 31;
  int ct   = (t >> 9)  &  3;
  int kc   =  t >> 11;
  int g = kc * 32 + ((lane & 16) ? 8 : 0) + ((j < 8) ? j : (j + 8));   // gaussian (K row)
  int c = ct * 16 + (lane & 15);                  // channel (N col)
  float x = feat[g * NCH + c];
  float sp = (x > 20.0f) ? x : log1pf(__expf(x)); // softplus
  bpack[t] = (_Float16)sp;
}

// ---------------------------------------------------------------------------
// Kernel 3: fused wmap generation + WMMA accumulation.
// 512 blocks x 256 threads (8 waves). Wave owns 16 row-aligned pixels;
// accumulates 16px x 64ch in 4 f32 WMMA tiles over 32 K-chunks.
// Per eval: 3x ds_load_b128 (AoS params, immediate offsets), 1 v_exp,
// 2 v_cos, ~14 VALU; per chunk: 4 v_wmma against pre-packed B.
// ---------------------------------------------------------------------------
__global__ __launch_bounds__(256) void gi_render(
    const float* __restrict__ P, const _Float16* __restrict__ bpack,
    float* __restrict__ out)
{
  __shared__ float sg[12 * N_G];                  // 48 KB AoS params

  int tid = threadIdx.x;
  {
    const float4* Pv = (const float4*)P;
    float4* sv = (float4*)sg;
#pragma unroll
    for (int it = 0; it < 12; ++it)               // 12288 floats / (256*4)
      sv[tid + it * 256] = Pv[tid + it * 256];
  }
  __syncthreads();

  int lane = tid & 31;
  int wave = tid >> 5;
  int m    = lane & 15;                           // pixel row within A tile
  int side = (lane & 16) ? 8 : 0;                 // A-operand K-side offset

  int pbase = blockIdx.x * 128 + wave * 16;       // 16 consecutive pixels
  int p = pbase + m;
  float px = (float)(p & (WDIM - 1)) + 0.5f;
  float py = (float)(p >> 8) + 0.5f;

  v8f acc0 = {}, acc1 = {}, acc2 = {}, acc3 = {};

  const v16h* bp = (const v16h*)bpack;
  const float4* sg4 = (const float4*)sg;          // 3 float4 per gaussian

  for (int kc = 0; kc < NCHUNK; ++kc) {
    int gbase = kc * 32 + side;                   // per-chunk LDS base

    AOp a;
#pragma unroll
    for (int j = 0; j < 16; ++j) {
      // A 16x32 f16 layout: lane<16 -> K in {0..7, 16..23}, lane>=16 -> +8
      int g = gbase + ((j < 8) ? j : (j + 8));    // const offset after unroll
      float4 q0 = sg4[g * 3 + 0];                 // xs  ys  ca_s cb_s
      float4 q1 = sg4[g * 3 + 1];                 // cc_s amp f0x  f0y
      float4 q2 = sg4[g * 3 + 2];                 // f1x f1y  w0   w1

      float dx = q0.x - px;
      float dy = q0.y - py;
      float t  = q0.z * (dx * dx) + q1.x * (dy * dy) + q0.w * (dx * dy);
      float e  = __builtin_amdgcn_exp2f(t);       // exp(-sigma)
      float c0 = __builtin_amdgcn_cosf(q1.z * dx + q1.w * dy);
      float c1 = __builtin_amdgcn_cosf(q2.x * dx + q2.y * dy);
      float mod = 1.0f + q2.z * c0 + q2.w * c1;
      a.e[j] = (_Float16)(q1.y * e * mod);
    }

    // 4 channel tiles: B operand = 32B contiguous per lane
    const v16h* bc = bp + kc * 128 + lane;
    v16h b0 = bc[0];
    v16h b1 = bc[32];
    v16h b2 = bc[64];
    v16h b3 = bc[96];

    acc0 = __builtin_amdgcn_wmma_f32_16x16x32_f16(false, a.v, false, b0,
                                                  (short)0, acc0, false, false);
    acc1 = __builtin_amdgcn_wmma_f32_16x16x32_f16(false, a.v, false, b1,
                                                  (short)0, acc1, false, false);
    acc2 = __builtin_amdgcn_wmma_f32_16x16x32_f16(false, a.v, false, b2,
                                                  (short)0, acc2, false, false);
    acc3 = __builtin_amdgcn_wmma_f32_16x16x32_f16(false, a.v, false, b3,
                                                  (short)0, acc3, false, false);
  }

  // D layout: lane holds channel N = lane&15, pixels M = side + r (r=0..7),
  // which are 8 *consecutive* pixels in channel-major output -> 2x b128.
  int prow = pbase + side;
#pragma unroll
  for (int ct = 0; ct < 4; ++ct) {
    v8f ac = (ct == 0) ? acc0 : (ct == 1) ? acc1 : (ct == 2) ? acc2 : acc3;
    int c = ct * 16 + m;
    float* op = out + (size_t)c * NPIX + prow;
    float4 lo, hi;
    lo.x = fminf(fmaxf(ac[0], 0.0f), 1.0f);
    lo.y = fminf(fmaxf(ac[1], 0.0f), 1.0f);
    lo.z = fminf(fmaxf(ac[2], 0.0f), 1.0f);
    lo.w = fminf(fmaxf(ac[3], 0.0f), 1.0f);
    hi.x = fminf(fmaxf(ac[4], 0.0f), 1.0f);
    hi.y = fminf(fmaxf(ac[5], 0.0f), 1.0f);
    hi.z = fminf(fmaxf(ac[6], 0.0f), 1.0f);
    hi.w = fminf(fmaxf(ac[7], 0.0f), 1.0f);
    *(float4*)(op)     = lo;
    *(float4*)(op + 4) = hi;
  }
}

// ---------------------------------------------------------------------------
extern "C" void kernel_launch(void* const* d_in, const int* in_sizes, int n_in,
                              void* d_out, int out_size, void* d_ws, size_t ws_size,
                              hipStream_t stream) {
  const float* xyz   = (const float*)d_in[0];  // (N,2)
  const float* chol  = (const float*)d_in[1];  // (N,3)
  const float* opac  = (const float*)d_in[2];  // (N,1)
  const float* feat  = (const float*)d_in[3];  // (N,64)
  const float* gfreq = (const float*)d_in[4];  // (N*2,2)
  const float* gwt   = (const float*)d_in[5];  // (N*2,1)

  float*    Pp    = (float*)d_ws;                              // 12*1024 f32 = 48 KB
  _Float16* bpack = (_Float16*)((char*)d_ws + 12 * N_G * 4);   // 65536 f16 = 128 KB
  float*    out   = (float*)d_out;                             // (64, 256, 256)

  gi_precompute<<<(N_G + 255) / 256, 256, 0, stream>>>(xyz, chol, opac, gfreq, gwt, Pp);
  gi_packB<<<NPIX / 256, 256, 0, stream>>>(feat, bpack);
  gi_render<<<NPIX / 128, 256, 0, stream>>>(Pp, bpack, out);
}
